// S5Layer_34205119545859
// MI455X (gfx1250) — matrix operand order repaired and checked by reading, not compiled, not run on hardware
//
#include <hip/hip_runtime.h>
#include <hip/hip_bf16.h>

// S5 layer: L=8192, H=1024, P=512, CONJ_SYM=true (mult=2)
#define L_SEQ 8192
#define H_DIM 1024
#define P_DIM 512

typedef __attribute__((ext_vector_type(16))) __bf16 v16bf;
typedef __attribute__((ext_vector_type(8)))  float  v8f;

union Frag { v16bf v; uint4 q[2]; };

__device__ __forceinline__ unsigned short f2bf(float f) {
  unsigned int u = __float_as_uint(f);
  u += 0x7FFFu + ((u >> 16) & 1u);        // round-to-nearest-even
  return (unsigned short)(u >> 16);
}

// One 16B async DMA: global -> LDS (ASYNCcnt-tracked, gfx1250).
// LDS offset = low 32 bits of the generic pointer to __shared__ memory.
__device__ __forceinline__ void async_b128(unsigned lds_off, const unsigned short* g) {
  asm volatile("global_load_async_to_lds_b128 %0, %1, off"
               :: "v"(lds_off), "v"(g) : "memory");
}

// ------------------------------------------------------------------
// Prep: Lambda_bar = exp(Lambda*step); coeff = (Lambda_bar-1)/Lambda
// ------------------------------------------------------------------
__global__ void prep_lambda(const float* __restrict__ Lre, const float* __restrict__ Lim,
                            const float* __restrict__ logstep,
                            float* __restrict__ lb, float* __restrict__ coeff) {
  int p = blockIdx.x * blockDim.x + threadIdx.x;
  if (p >= P_DIM) return;
  float lr = Lre[p], li = Lim[p];
  float s  = expf(logstep[p]);
  float ar = lr * s, ai = li * s;
  float e  = expf(ar);
  float lbr = e * cosf(ai), lbi = e * sinf(ai);
  lb[2*p] = lbr; lb[2*p+1] = lbi;
  float dr = lbr - 1.0f, di = lbi;
  float inv = 1.0f / (lr*lr + li*li);
  coeff[2*p]   = (dr*lr + di*li) * inv;
  coeff[2*p+1] = (di*lr - dr*li) * inv;
}

// B_bar[p,h] = coeff[p] * B_tilde[p,h]  -> bf16 (P,H) row-major (re, im)
__global__ void prep_weights(const float* __restrict__ B, const float* __restrict__ coeff,
                             unsigned short* __restrict__ Wre, unsigned short* __restrict__ Wim) {
  int idx = blockIdx.x * blockDim.x + threadIdx.x;
  if (idx >= P_DIM * H_DIM) return;
  int p = idx / H_DIM;
  float br = B[(size_t)idx * 2], bi = B[(size_t)idx * 2 + 1];
  float cr = coeff[2*p], ci = coeff[2*p+1];
  Wre[idx] = f2bf(cr*br - ci*bi);
  Wim[idx] = f2bf(cr*bi + ci*br);
}

// C_tilde (H,P,2) -> bf16 Cre (H,P), Cimneg = -Cim (H,P)
__global__ void prep_c(const float* __restrict__ C,
                       unsigned short* __restrict__ Cre, unsigned short* __restrict__ Cin) {
  int idx = blockIdx.x * blockDim.x + threadIdx.x;
  if (idx >= H_DIM * P_DIM) return;
  Cre[idx] = f2bf(C[(size_t)idx * 2]);
  Cin[idx] = f2bf(-C[(size_t)idx * 2 + 1]);
}

// U (L,H) fp32 -> bf16
__global__ void prep_u(const float* __restrict__ U, unsigned short* __restrict__ Ubf) {
  int idx = blockIdx.x * blockDim.x + threadIdx.x;
  if (idx < L_SEQ * H_DIM) Ubf[idx] = f2bf(U[idx]);
}

// ------------------------------------------------------------------
// GEMM tiling: WG tile 64(M) x 128(N), K-chunk 32, 8 waves (2M x 4N).
// LDS rows padded to 40 elements (80B) -> conflict-free ds_load_b128.
// ------------------------------------------------------------------
#define KC 32
#define LDSTR 40

// GEMM1: Bu(L,P) = U(L,H) @ B_bar^T  (complex; two f32 outputs)
__global__ __launch_bounds__(256) void gemm_bu(const unsigned short* __restrict__ Ubf,
                                               const unsigned short* __restrict__ Wre,
                                               const unsigned short* __restrict__ Wim,
                                               float* __restrict__ BuRe,
                                               float* __restrict__ BuIm) {
  __shared__ __align__(16) unsigned short sA [2][64  * LDSTR];
  __shared__ __align__(16) unsigned short sBr[2][128 * LDSTR];
  __shared__ __align__(16) unsigned short sBi[2][128 * LDSTR];

  const int tid   = threadIdx.x;
  const int lane  = tid & 31;
  const int wave  = tid >> 5;
  const int ln16  = lane & 15;
  const int khalf = lane >> 4;
  const int wm = wave & 1, wn = wave >> 1;
  const int mWG = blockIdx.x * 64;
  const int nWG = blockIdx.y * 128;

  // per-thread 16B copy slots (4 chunks per 32-elem row)
  const int rA  = tid >> 2,          cA  = (tid & 3) * 8;
  const int rB0 = tid >> 2,          cB0 = (tid & 3) * 8;
  const int rB1 = (tid + 256) >> 2,  cB1 = ((tid + 256) & 3) * 8;

  const unsigned aOff  = (unsigned)(size_t)(const void*)&sA[0][0];
  const unsigned brOff = (unsigned)(size_t)(const void*)&sBr[0][0];
  const unsigned biOff = (unsigned)(size_t)(const void*)&sBi[0][0];
  const unsigned bufA = sizeof(sA[0]), bufB = sizeof(sBr[0]);

  auto stage = [&](int buf, int k0) {   // 5 async b128 per thread
    async_b128(aOff  + buf*bufA + (unsigned)(rA*LDSTR + cA)*2,
               Ubf + (size_t)(mWG + rA)*H_DIM + k0 + cA);
    async_b128(brOff + buf*bufB + (unsigned)(rB0*LDSTR + cB0)*2,
               Wre + (size_t)(nWG + rB0)*H_DIM + k0 + cB0);
    async_b128(brOff + buf*bufB + (unsigned)(rB1*LDSTR + cB1)*2,
               Wre + (size_t)(nWG + rB1)*H_DIM + k0 + cB1);
    async_b128(biOff + buf*bufB + (unsigned)(rB0*LDSTR + cB0)*2,
               Wim + (size_t)(nWG + rB0)*H_DIM + k0 + cB0);
    async_b128(biOff + buf*bufB + (unsigned)(rB1*LDSTR + cB1)*2,
               Wim + (size_t)(nWG + rB1)*H_DIM + k0 + cB1);
  };

  v8f accR[2][2] = {};
  v8f accI[2][2] = {};

  auto compute = [&](int b) {           // called with literal 0/1 only
    Frag a[2], br[2], bi[2];
#pragma unroll
    for (int mi = 0; mi < 2; ++mi) {
      const unsigned short* p = &sA[b][(wm*32 + mi*16 + ln16) * LDSTR + khalf*8];
      a[mi].q[0] = *(const uint4*)p;
      a[mi].q[1] = *(const uint4*)(p + 16);
    }
#pragma unroll
    for (int ni = 0; ni < 2; ++ni) {
      const unsigned short* pr = &sBr[b][(wn*32 + ni*16 + ln16) * LDSTR + khalf*8];
      br[ni].q[0] = *(const uint4*)pr;
      br[ni].q[1] = *(const uint4*)(pr + 16);
      const unsigned short* pi = &sBi[b][(wn*32 + ni*16 + ln16) * LDSTR + khalf*8];
      bi[ni].q[0] = *(const uint4*)pi;
      bi[ni].q[1] = *(const uint4*)(pi + 16);
    }
#pragma unroll
    for (int mi = 0; mi < 2; ++mi)
#pragma unroll
      for (int ni = 0; ni < 2; ++ni) {
        accR[mi][ni] = __builtin_amdgcn_wmma_f32_16x16x32_bf16(
            false, a[mi].v, false, br[ni].v, (short)0, accR[mi][ni], false, false);
        accI[mi][ni] = __builtin_amdgcn_wmma_f32_16x16x32_bf16(
            false, a[mi].v, false, bi[ni].v, (short)0, accI[mi][ni], false, false);
      }
  };

  const int NK = H_DIM / KC;            // 32 (even)
  stage(0, 0);
  stage(1, KC);
#pragma unroll 1
  for (int k = 0; k < NK; k += 2) {
    // ---- step k: buffer 0 ----
    asm volatile("s_wait_asynccnt 0x5" ::: "memory");   // batch k landed
    __syncthreads();
    compute(0);
    __syncthreads();
    if (k + 2 < NK) stage(0, (k + 2) * KC);
    // ---- step k+1: buffer 1 ----
    if (k + 2 < NK) asm volatile("s_wait_asynccnt 0x5" ::: "memory");
    else            asm volatile("s_wait_asynccnt 0x0" ::: "memory");
    __syncthreads();
    compute(1);
    __syncthreads();
    if (k + 3 < NK) stage(1, (k + 3) * KC);
  }

  const int mBase = mWG + wm*32, nBase = nWG + wn*32;
#pragma unroll
  for (int mi = 0; mi < 2; ++mi)
#pragma unroll
    for (int ni = 0; ni < 2; ++ni)
#pragma unroll
      for (int r = 0; r < 8; ++r) {
        int row = mBase + mi*16 + khalf*8 + r;
        int col = nBase + ni*16 + ln16;
        BuRe[(size_t)row * P_DIM + col] = accR[mi][ni][r];
        BuIm[(size_t)row * P_DIM + col] = accI[mi][ni][r];
      }
}

// ------------------------------------------------------------------
// Chunked associative scan: x[l] = lb*x[l-1] + Bu[l]  (complex, per p)
// ------------------------------------------------------------------
#define SC_P 4
#define SC_C 64
#define SC_LEN (L_SEQ / SC_C)   // 128

__global__ __launch_bounds__(256) void scan_states(const float* __restrict__ lb,
                                                   const float* __restrict__ BuRe,
                                                   const float* __restrict__ BuIm,
                                                   unsigned short* __restrict__ Xre,
                                                   unsigned short* __restrict__ Xim,
                                                   float* __restrict__ finalState) {
  __shared__ float sAr[SC_C][SC_P], sAi[SC_C][SC_P];
  __shared__ float sbr[SC_C][SC_P], sbi[SC_C][SC_P];

  const int pl = threadIdx.x & (SC_P - 1);
  const int c  = threadIdx.x / SC_P;
  const int p  = blockIdx.x * SC_P + pl;
  const int l0 = c * SC_LEN;

  const float lbr = lb[2*p], lbi = lb[2*p+1];

  float Ar = lbr, Ai = lbi;              // lb^128 via 7 complex squarings
#pragma unroll
  for (int i = 0; i < 7; ++i) {
    float nr = Ar*Ar - Ai*Ai;
    Ai = 2.0f * Ar * Ai;
    Ar = nr;
  }

  float xr = 0.0f, xi = 0.0f;            // pass 1: chunk-local reduction
  for (int i = 0; i < SC_LEN; ++i) {
    size_t off = (size_t)(l0 + i) * P_DIM + p;
    float br = BuRe[off], bi = BuIm[off];
    float t = lbr*xr - lbi*xi + br;
    xi      = lbr*xi + lbi*xr + bi;
    xr = t;
  }

  sAr[c][pl] = Ar; sAi[c][pl] = Ai; sbr[c][pl] = xr; sbi[c][pl] = xi;
  __syncthreads();
  for (int s = 1; s < SC_C; s <<= 1) {   // Hillis-Steele on (A,b)
    float pAr = 0.f, pAi = 0.f, pbr = 0.f, pbi = 0.f;
    const bool has = (c >= s);
    if (has) { pAr = sAr[c-s][pl]; pAi = sAi[c-s][pl];
               pbr = sbr[c-s][pl]; pbi = sbi[c-s][pl]; }
    __syncthreads();
    if (has) {
      float nAr = Ar*pAr - Ai*pAi;
      float nAi = Ar*pAi + Ai*pAr;
      float nbr = Ar*pbr - Ai*pbi + xr;
      float nbi = Ar*pbi + Ai*pbr + xi;
      Ar = nAr; Ai = nAi; xr = nbr; xi = nbi;
      sAr[c][pl] = Ar; sAi[c][pl] = Ai; sbr[c][pl] = xr; sbi[c][pl] = xi;
    }
    __syncthreads();
  }

  float cr = 0.0f, ci = 0.0f;            // carry = inclusive scan of chunk c-1
  if (c > 0) { cr = sbr[c-1][pl]; ci = sbi[c-1][pl]; }

  xr = cr; xi = ci;                      // pass 2: replay + emit bf16 states
  for (int i = 0; i < SC_LEN; ++i) {
    size_t off = (size_t)(l0 + i) * P_DIM + p;
    float br = BuRe[off], bi = BuIm[off];
    float t = lbr*xr - lbi*xi + br;
    xi      = lbr*xi + lbi*xr + bi;
    xr = t;
    Xre[off] = f2bf(xr);
    Xim[off] = f2bf(xi);
  }
  if (c == SC_C - 1) {                   // x[L-1] -> final state (complex64)
    finalState[2*p]   = xr;
    finalState[2*p+1] = xi;
  }
}

// ------------------------------------------------------------------
// GEMM2: out(L,H) = 2*(Xre@Cre^T + Xim@(-Cim)^T) + D*U
// ------------------------------------------------------------------
__global__ __launch_bounds__(256) void gemm_out(const unsigned short* __restrict__ Xre,
                                                const unsigned short* __restrict__ Xim,
                                                const unsigned short* __restrict__ Cre,
                                                const unsigned short* __restrict__ Cin,
                                                const float* __restrict__ U,
                                                const float* __restrict__ Dvec,
                                                float* __restrict__ out) {
  __shared__ __align__(16) unsigned short sXr[2][64  * LDSTR];
  __shared__ __align__(16) unsigned short sXi[2][64  * LDSTR];
  __shared__ __align__(16) unsigned short sCr[2][128 * LDSTR];
  __shared__ __align__(16) unsigned short sCi[2][128 * LDSTR];

  const int tid   = threadIdx.x;
  const int lane  = tid & 31;
  const int wave  = tid >> 5;
  const int ln16  = lane & 15;
  const int khalf = lane >> 4;
  const int wm = wave & 1, wn = wave >> 1;
  const int mWG = blockIdx.x * 64;
  const int nWG = blockIdx.y * 128;

  const int rA  = tid >> 2,          cA  = (tid & 3) * 8;
  const int rB0 = tid >> 2,          cB0 = (tid & 3) * 8;
  const int rB1 = (tid + 256) >> 2,  cB1 = ((tid + 256) & 3) * 8;

  const unsigned xrOff = (unsigned)(size_t)(const void*)&sXr[0][0];
  const unsigned xiOff = (unsigned)(size_t)(const void*)&sXi[0][0];
  const unsigned crOff = (unsigned)(size_t)(const void*)&sCr[0][0];
  const unsigned ciOff = (unsigned)(size_t)(const void*)&sCi[0][0];
  const unsigned bufA = sizeof(sXr[0]), bufB = sizeof(sCr[0]);

  auto stage = [&](int buf, int k0) {   // 6 async b128 per thread
    async_b128(xrOff + buf*bufA + (unsigned)(rA*LDSTR + cA)*2,
               Xre + (size_t)(mWG + rA)*P_DIM + k0 + cA);
    async_b128(xiOff + buf*bufA + (unsigned)(rA*LDSTR + cA)*2,
               Xim + (size_t)(mWG + rA)*P_DIM + k0 + cA);
    async_b128(crOff + buf*bufB + (unsigned)(rB0*LDSTR + cB0)*2,
               Cre + (size_t)(nWG + rB0)*P_DIM + k0 + cB0);
    async_b128(crOff + buf*bufB + (unsigned)(rB1*LDSTR + cB1)*2,
               Cre + (size_t)(nWG + rB1)*P_DIM + k0 + cB1);
    async_b128(ciOff + buf*bufB + (unsigned)(rB0*LDSTR + cB0)*2,
               Cin + (size_t)(nWG + rB0)*P_DIM + k0 + cB0);
    async_b128(ciOff + buf*bufB + (unsigned)(rB1*LDSTR + cB1)*2,
               Cin + (size_t)(nWG + rB1)*P_DIM + k0 + cB1);
  };

  v8f acc[2][2] = {};

  auto compute = [&](int b) {           // called with literal 0/1 only
    Frag ar[2], ai[2], br[2], bi[2];
#pragma unroll
    for (int mi = 0; mi < 2; ++mi) {
      const unsigned short* pr = &sXr[b][(wm*32 + mi*16 + ln16) * LDSTR + khalf*8];
      ar[mi].q[0] = *(const uint4*)pr;
      ar[mi].q[1] = *(const uint4*)(pr + 16);
      const unsigned short* pi = &sXi[b][(wm*32 + mi*16 + ln16) * LDSTR + khalf*8];
      ai[mi].q[0] = *(const uint4*)pi;
      ai[mi].q[1] = *(const uint4*)(pi + 16);
    }
#pragma unroll
    for (int ni = 0; ni < 2; ++ni) {
      const unsigned short* pr = &sCr[b][(wn*32 + ni*16 + ln16) * LDSTR + khalf*8];
      br[ni].q[0] = *(const uint4*)pr;
      br[ni].q[1] = *(const uint4*)(pr + 16);
      const unsigned short* pi = &sCi[b][(wn*32 + ni*16 + ln16) * LDSTR + khalf*8];
      bi[ni].q[0] = *(const uint4*)pi;
      bi[ni].q[1] = *(const uint4*)(pi + 16);
    }
    // two sweeps: dependent WMMAs on the same accumulator are 4 apart
#pragma unroll
    for (int mi = 0; mi < 2; ++mi)
#pragma unroll
      for (int ni = 0; ni < 2; ++ni)
        acc[mi][ni] = __builtin_amdgcn_wmma_f32_16x16x32_bf16(
            false, ar[mi].v, false, br[ni].v, (short)0, acc[mi][ni], false, false);
#pragma unroll
    for (int mi = 0; mi < 2; ++mi)
#pragma unroll
      for (int ni = 0; ni < 2; ++ni)
        acc[mi][ni] = __builtin_amdgcn_wmma_f32_16x16x32_bf16(
            false, ai[mi].v, false, bi[ni].v, (short)0, acc[mi][ni], false, false);
  };

  const int NK = P_DIM / KC;            // 16 (even)
  stage(0, 0);
  stage(1, KC);
#pragma unroll 1
  for (int k = 0; k < NK; k += 2) {
    asm volatile("s_wait_asynccnt 0x6" ::: "memory");
    __syncthreads();
    compute(0);
    __syncthreads();
    if (k + 2 < NK) stage(0, (k + 2) * KC);
    if (k + 2 < NK) asm volatile("s_wait_asynccnt 0x6" ::: "memory");
    else            asm volatile("s_wait_asynccnt 0x0" ::: "memory");
    __syncthreads();
    compute(1);
    __syncthreads();
    if (k + 3 < NK) stage(1, (k + 3) * KC);
  }

  const int mBase = mWG + wm*32, nBase = nWG + wn*32;
  float dv[2];
#pragma unroll
  for (int ni = 0; ni < 2; ++ni) dv[ni] = Dvec[nBase + ni*16 + ln16];

#pragma unroll
  for (int mi = 0; mi < 2; ++mi)
#pragma unroll
    for (int ni = 0; ni < 2; ++ni)
#pragma unroll
      for (int r = 0; r < 8; ++r) {
        int l = mBase + mi*16 + khalf*8 + r;
        int h = nBase + ni*16 + ln16;
        size_t off = (size_t)l * H_DIM + h;
        out[off] = 2.0f * acc[mi][ni][r] + dv[ni] * U[off];
      }
}

// ------------------------------------------------------------------
extern "C" void kernel_launch(void* const* d_in, const int* in_sizes, int n_in,
                              void* d_out, int out_size, void* d_ws, size_t ws_size,
                              hipStream_t stream) {
  const float* U       = (const float*)d_in[0];   // (L,H)
  const float* Lre     = (const float*)d_in[1];   // (P,)
  const float* Lim     = (const float*)d_in[2];   // (P,)
  const float* B       = (const float*)d_in[3];   // (P,H,2)
  const float* C       = (const float*)d_in[4];   // (H,P,2)
  const float* Dv      = (const float*)d_in[5];   // (H,)
  const float* logstep = (const float*)d_in[6];   // (P,)
  float* out = (float*)d_out;                     // (L,H) ++ final state (2P floats)

  char* w = (char*)d_ws;
  float* lb    = (float*)(w);                         // 2P
  float* coeff = (float*)(w + 4096);                  // 2P
  unsigned short* Wre = (unsigned short*)(w + 8192);  // P*H bf16
  unsigned short* Wim = Wre + (size_t)P_DIM * H_DIM;
  unsigned short* Cre = Wim + (size_t)P_DIM * H_DIM;  // H*P bf16
  unsigned short* Cin = Cre + (size_t)H_DIM * P_DIM;
  unsigned short* Ubf = Cin + (size_t)H_DIM * P_DIM;  // L*H bf16
  float* BuRe = (float*)(Ubf + (size_t)L_SEQ * H_DIM);        // L*P f32
  float* BuIm = BuRe + (size_t)L_SEQ * P_DIM;
  unsigned short* Xre = (unsigned short*)(BuIm + (size_t)L_SEQ * P_DIM); // L*P bf16
  unsigned short* Xim = Xre + (size_t)L_SEQ * P_DIM;

  prep_lambda<<<(P_DIM + 255) / 256, 256, 0, stream>>>(Lre, Lim, logstep, lb, coeff);
  prep_weights<<<(P_DIM * H_DIM + 255) / 256, 256, 0, stream>>>(B, coeff, Wre, Wim);
  prep_c<<<(H_DIM * P_DIM + 255) / 256, 256, 0, stream>>>(C, Cre, Cin);
  prep_u<<<(L_SEQ * H_DIM + 255) / 256, 256, 0, stream>>>(U, Ubf);

  gemm_bu<<<dim3(L_SEQ / 64, P_DIM / 128), 256, 0, stream>>>(Ubf, Wre, Wim, BuRe, BuIm);

  scan_states<<<P_DIM / SC_P, SC_P * SC_C, 0, stream>>>(
      lb, BuRe, BuIm, Xre, Xim, out + (size_t)L_SEQ * H_DIM);

  gemm_out<<<dim3(L_SEQ / 64, H_DIM / 128), 256, 0, stream>>>(
      Xre, Xim, Cre, Cin, U, Dv, out);
}
